// MultiHeadAttention_36627481100800
// MI455X (gfx1250) — matrix operand order repaired and checked by reading, not compiled, not run on hardware
//
#include <hip/hip_runtime.h>
#include <hip/hip_bf16.h>

#define DI __device__ __forceinline__

typedef unsigned int   u32;
typedef unsigned short u16;
typedef __bf16 bf16;
typedef bf16  bf16x16 __attribute__((ext_vector_type(16)));
typedef float f32x8   __attribute__((ext_vector_type(8)));

static constexpr int Bc = 4, Sc = 2048, Ec = 1024, Hc = 16, HDc = 64;
static constexpr int Mtot = Bc * Sc; // 8192

union FragB { uint4 q[2]; bf16x16 v; };

DI u16 f2bf(float f) {   // fp32 -> bf16, round-half-up (ties differ from RNE)
  const u32 u = __builtin_bit_cast(u32, f);
  return (u16)((u + 0x8000u) >> 16);
}
#if __has_builtin(__builtin_amdgcn_perm)
DI u32 pack2(float a, float b) {             // 2 adds + 1 v_perm_b32
  const u32 au = __builtin_bit_cast(u32, a) + 0x8000u;
  const u32 bu = __builtin_bit_cast(u32, b) + 0x8000u;
  return __builtin_amdgcn_perm(bu, au, 0x07060302u); // {bf(b), bf(a)}
}
#else
DI u32 pack2(float a, float b) { return (u32)f2bf(a) | ((u32)f2bf(b) << 16); }
#endif

DI f32x8 wmma_bf16(const FragB& a, const FragB& b, f32x8 c) {
  return __builtin_amdgcn_wmma_f32_16x16x32_bf16(
      /*neg_a=*/false, a.v, /*neg_b=*/false, b.v,
      /*c_mod=*/(short)0, c, /*reuse_a=*/false, /*reuse_b=*/false);
}

// --- gfx1250 async global->LDS staging (ASYNCcnt path), with fallback -------
#if defined(__AMDGCN__) && __has_builtin(__builtin_amdgcn_global_load_async_to_lds_b128)
#define HAVE_ASYNC_LDS 1
typedef int i32x4v __attribute__((vector_size(16)));
typedef __attribute__((address_space(1))) i32x4v gv4i;
typedef __attribute__((address_space(3))) i32x4v lv4i;
#else
#define HAVE_ASYNC_LDS 0
#endif

DI void stage16(const void* g, void* l) {    // copy 16B global -> LDS
#if HAVE_ASYNC_LDS
  __builtin_amdgcn_global_load_async_to_lds_b128((gv4i*)g, (lv4i*)l, 0, 0);
#else
  *(uint4*)l = *(const uint4*)g;
#endif
}
DI void stage_wait() {                       // drain ASYNCcnt
#if HAVE_ASYNC_LDS
#if __has_builtin(__builtin_amdgcn_s_wait_asynccnt)
  __builtin_amdgcn_s_wait_asynccnt(0);
#else
  asm volatile("s_wait_asynccnt 0x0" ::: "memory");
#endif
#endif
}

// ---------------------------------------------------------------------------
// Weight transpose + convert: Wt[n][k] = bf16(W[k][n]), E x E
// ---------------------------------------------------------------------------
__global__ void wconv_t(const float* __restrict__ W, bf16* __restrict__ Wt) {
  __shared__ float tile[32][33];
  const int tx = threadIdx.x, ty = threadIdx.y;
  const int n0 = blockIdx.x * 32, k0 = blockIdx.y * 32;
#pragma unroll
  for (int i = 0; i < 32; i += 8)
    tile[ty + i][tx] = W[(size_t)(k0 + ty + i) * Ec + n0 + tx];
  __syncthreads();
#pragma unroll
  for (int i = 0; i < 32; i += 8)
    *(u16*)&Wt[(size_t)(n0 + ty + i) * Ec + k0 + tx] = f2bf(tile[tx][ty + i]);
}

// ---------------------------------------------------------------------------
// Tiled GEMM: Y[m][n] = sum_k A[m][k] * Wt[n][k] + bias[n]  (M=8192, N=K=1024)
// Double-buffered LDS tiles; next tile's async DMA issued before computing
// on the current one. MODE 0: bf16 head-split out; MODE 1: bf16 transposed
// head out; MODE 2: fp32 flat out. AF32: fp32 A converted during staging.
// ---------------------------------------------------------------------------
template <int MODE, bool AF32>
__global__ __launch_bounds__(256) void gemm_qkv(
    const void* __restrict__ Aptr, const bf16* __restrict__ Wt,
    const float* __restrict__ bias, void* __restrict__ Out, float scale)
{
  constexpr int BK = 32, LDT = 40, TSZ = 128 * LDT;  // padded stride (halves)
  __shared__ __align__(16) bf16 As[2 * TSZ];
  __shared__ __align__(16) bf16 Bs[2 * TSZ];

  const int t = threadIdx.x;
  const int lane = t & 31, wid = t >> 5;
  const int wm = wid >> 2, wn = wid & 3;     // wave grid 2 (M) x 4 (N)
  const int bn = blockIdx.x * 128, bm = blockIdx.y * 128;
  const int nlo = lane & 15, lhalf = lane >> 4;
  const int srow = t >> 1, sseg = t & 1;     // staging: 2 threads / row, 32B

  auto stage_tiles = [&](int k0, int buf) {
    bf16* Asb = As + buf * TSZ;
    bf16* Bsb = Bs + buf * TSZ;
    stage16(Wt + (size_t)(bn + srow) * Ec + k0 + sseg * 16,
            &Bsb[srow * LDT + sseg * 16]);
    stage16(Wt + (size_t)(bn + srow) * Ec + k0 + sseg * 16 + 8,
            &Bsb[srow * LDT + sseg * 16 + 8]);
    if (AF32) {                              // fp32 A: convert while staging
      const float* A = (const float*)Aptr;
      const float4* src =
          (const float4*)(A + (size_t)(bm + srow) * Ec + k0 + sseg * 16);
      float4 f0 = src[0], f1 = src[1], f2 = src[2], f3 = src[3];
      uint4 p0, p1;
      p0.x = pack2(f0.x, f0.y); p0.y = pack2(f0.z, f0.w);
      p0.z = pack2(f1.x, f1.y); p0.w = pack2(f1.z, f1.w);
      p1.x = pack2(f2.x, f2.y); p1.y = pack2(f2.z, f2.w);
      p1.z = pack2(f3.x, f3.y); p1.w = pack2(f3.z, f3.w);
      *(uint4*)&Asb[srow * LDT + sseg * 16]     = p0;
      *(uint4*)&Asb[srow * LDT + sseg * 16 + 8] = p1;
    } else {                                 // bf16 A: async stage
      const bf16* A = (const bf16*)Aptr;
      stage16(A + (size_t)(bm + srow) * Ec + k0 + sseg * 16,
              &Asb[srow * LDT + sseg * 16]);
      stage16(A + (size_t)(bm + srow) * Ec + k0 + sseg * 16 + 8,
              &Asb[srow * LDT + sseg * 16 + 8]);
    }
  };

  f32x8 acc[4][2];
#pragma unroll
  for (int i = 0; i < 4; i++)
#pragma unroll
    for (int j = 0; j < 2; j++)
#pragma unroll
      for (int r = 0; r < 8; r++) acc[i][j][r] = 0.f;

  constexpr int NK = Ec / BK;                // 32 k-steps
  stage_tiles(0, 0);
  for (int kt = 0; kt < NK; kt++) {
    stage_wait();
    __syncthreads();                         // buf[kt&1] ready; buf[(kt+1)&1] free
    if (kt + 1 < NK) stage_tiles((kt + 1) * BK, (kt + 1) & 1);
    const bf16* Asb = As + (kt & 1) * TSZ;
    const bf16* Bsb = Bs + (kt & 1) * TSZ;

    FragB bfr[2];
#pragma unroll
    for (int j = 0; j < 2; j++) {
      const int n = wn * 32 + j * 16 + nlo;
      const int kb = lhalf * 8;
      bfr[j].q[0] = *(const uint4*)&Bsb[n * LDT + kb];
      bfr[j].q[1] = *(const uint4*)&Bsb[n * LDT + kb + 16];
    }
#pragma unroll
    for (int i = 0; i < 4; i++) {
      FragB afr;
      const int r = wm * 64 + i * 16 + nlo;
      const int kb = lhalf * 8;
      afr.q[0] = *(const uint4*)&Asb[r * LDT + kb];
      afr.q[1] = *(const uint4*)&Asb[r * LDT + kb + 16];
#pragma unroll
      for (int j = 0; j < 2; j++) acc[i][j] = wmma_bf16(afr, bfr[j], acc[i][j]);
    }
  }

  // Epilogue (bf16 modes convert in pairs via v_perm packing)
#pragma unroll
  for (int i = 0; i < 4; i++) {
#pragma unroll
    for (int j = 0; j < 2; j++) {
      const int n = bn + wn * 32 + j * 16 + nlo;
      const float bv = bias[n];
#pragma unroll
      for (int r = 0; r < 8; r += 2) {
        const int m0 = bm + wm * 64 + i * 16 + r + lhalf * 8;
        const float v0 = (acc[i][j][r]     + bv) * scale;
        const float v1 = (acc[i][j][r + 1] + bv) * scale;
        if (MODE == 2) {
          ((float*)Out)[(size_t)m0 * Ec + n]       = v0;
          ((float*)Out)[(size_t)(m0 + 1) * Ec + n] = v1;
        } else {
          const u32 pk = pack2(v0, v1);
          const int bI = m0 >> 11, s = m0 & (Sc - 1), h = n >> 6, d = n & 63;
          bf16* ob = (bf16*)Out;
          if (MODE == 0) {
            const size_t o = (((size_t)(bI * Hc + h)) * Sc + s) * HDc + d;
            *(u16*)&ob[o]       = (u16)pk;
            *(u16*)&ob[o + HDc] = (u16)(pk >> 16);   // next s -> +HDc
          } else {
            const size_t o = (((size_t)(bI * Hc + h)) * HDc + d) * Sc + s;
            *(u16*)&ob[o]     = (u16)pk;
            *(u16*)&ob[o + 1] = (u16)(pk >> 16);     // next s -> +1
          }
        }
      }
    }
  }
}

// ---------------------------------------------------------------------------
// Fused flash attention. Q,K: [B*H][S][HD] bf16 (Q pre-scaled 0.125*log2e),
// Vt: [B*H][HD][S] bf16. X out: [B][S][E] bf16 (heads merged).
// Max-free softmax in exp2 domain; denominator via P x ones WMMA; KV tiles
// double-buffered through the async DMA engine.
// ---------------------------------------------------------------------------
__global__ __launch_bounds__(256) void attn_fwd(
    const bf16* __restrict__ Q, const bf16* __restrict__ K,
    const bf16* __restrict__ Vt, bf16* __restrict__ X)
{
  constexpr int LKS = 72, LVS = 40, LPS = 40;   // padded strides (halves)
  constexpr int KSZ = 32 * LKS, VSZ = 64 * LVS;
  __shared__ __align__(16) bf16 Ks[2 * KSZ];
  __shared__ __align__(16) bf16 Vs[2 * VSZ];
  __shared__ __align__(16) bf16 Ps[8 * 16 * LPS];

  const int t = threadIdx.x, lane = t & 31, wid = t >> 5;
  const int bh = blockIdx.x >> 4, qt = blockIdx.x & 15;
  const int bI = bh >> 4, h = bh & 15;
  const bf16* qp = Q  + (size_t)bh * Sc * HDc;
  const bf16* kp = K  + (size_t)bh * Sc * HDc;
  const bf16* vp = Vt + (size_t)bh * HDc * Sc;
  const int q0 = qt * 128 + wid * 16;
  const int nlo = lane & 15, lhalf = lane >> 4;

  auto stage_kv = [&](int c0, int buf) {
    if (t < 128) {                          // stage K tile: 32 rows x 64
      const int r = t >> 2, seg = t & 3;
      const bf16* src = kp + (size_t)(c0 + r) * HDc + seg * 16;
      bf16* dst = &Ks[buf * KSZ + r * LKS + seg * 16];
      stage16(src, dst);
      stage16(src + 8, dst + 8);
    } else {                                // stage Vt tile: 64 rows x 32
      const int t2 = t - 128, d = t2 >> 1, seg = t2 & 1;
      const bf16* src = vp + (size_t)d * Sc + c0 + seg * 16;
      bf16* dst = &Vs[buf * VSZ + d * LVS + seg * 16];
      stage16(src, dst);
      stage16(src + 8, dst + 8);
    }
  };

  // Persistent Q A-fragments (16 rows x 64 = two k=32 fragments)
  FragB qa[2];
  {
    const bf16* rp = qp + (size_t)(q0 + nlo) * HDc + lhalf * 8;
    qa[0].q[0] = *(const uint4*)(rp);
    qa[0].q[1] = *(const uint4*)(rp + 16);
    qa[1].q[0] = *(const uint4*)(rp + 32);
    qa[1].q[1] = *(const uint4*)(rp + 48);
  }
  FragB ones;                                // bf16 1.0 splat for row sums
  {
    const u32 one2 = 0x3F803F80u;
    ones.q[0] = make_uint4(one2, one2, one2, one2);
    ones.q[1] = make_uint4(one2, one2, one2, one2);
  }

  f32x8 O[4], Lacc;
#pragma unroll
  for (int dt = 0; dt < 4; dt++)
#pragma unroll
    for (int r = 0; r < 8; r++) O[dt][r] = 0.f;
#pragma unroll
  for (int r = 0; r < 8; r++) Lacc[r] = 0.f;

  constexpr int NIT = Sc / 32;               // 64 KV tiles
  stage_kv(0, 0);
  for (int it = 0; it < NIT; it++) {
    stage_wait();
    __syncthreads();                         // buf[it&1] ready; other free
    if (it + 1 < NIT) stage_kv((it + 1) * 32, (it + 1) & 1);
    const bf16* Ksb = Ks + (it & 1) * KSZ;
    const bf16* Vsb = Vs + (it & 1) * VSZ;

    // Scores: two 16x16 tiles over 32 keys, K-dim = 64 (2 wmma each)
    f32x8 s0, s1;
#pragma unroll
    for (int r = 0; r < 8; r++) { s0[r] = 0.f; s1[r] = 0.f; }
#pragma unroll
    for (int ks = 0; ks < 2; ks++) {
      FragB bk0, bk1;
      const int kb = ks * 32 + lhalf * 8;
      bk0.q[0] = *(const uint4*)&Ksb[(nlo)      * LKS + kb];
      bk0.q[1] = *(const uint4*)&Ksb[(nlo)      * LKS + kb + 16];
      bk1.q[0] = *(const uint4*)&Ksb[(16 + nlo) * LKS + kb];
      bk1.q[1] = *(const uint4*)&Ksb[(16 + nlo) * LKS + kb + 16];
      s0 = wmma_bf16(qa[ks], bk0, s0);
      s1 = wmma_bf16(qa[ks], bk1, s1);
    }

    // P = exp2(scores); pack pairs with v_perm, spill to per-wave LDS
    const int pbase = wid * 16 * LPS;
#pragma unroll
    for (int r = 0; r < 8; r++) {
      const float p0 = __builtin_amdgcn_exp2f(s0[r]);
      const float p1 = __builtin_amdgcn_exp2f(s1[r]);
      const u32 pk = pack2(p0, p1);
      const int prow = pbase + (r + lhalf * 8) * LPS + nlo;
      *(u16*)&Ps[prow]      = (u16)pk;
      *(u16*)&Ps[prow + 16] = (u16)(pk >> 16);
    }
    asm volatile("s_wait_dscnt 0x0" ::: "memory"); // wave's P stores visible

    FragB pf;                               // reload P in A-fragment layout
    {
      const int pr = pbase + nlo * LPS + lhalf * 8;
      pf.q[0] = *(const uint4*)&Ps[pr];
      pf.q[1] = *(const uint4*)&Ps[pr + 16];
    }
#pragma unroll
    for (int dt = 0; dt < 4; dt++) {        // O += P x V  (4 d-tiles)
      FragB bv;
      const int vr = (dt * 16 + nlo) * LVS + lhalf * 8;
      bv.q[0] = *(const uint4*)&Vsb[vr];
      bv.q[1] = *(const uint4*)&Vsb[vr + 16];
      O[dt] = wmma_bf16(pf, bv, O[dt]);
    }
    Lacc = wmma_bf16(pf, ones, Lacc);       // row sums via matrix pipe
  }

  // Normalize and store merged-head X (pair conversions)
#pragma unroll
  for (int r = 0; r < 8; r++) {
    const float inv = __builtin_amdgcn_rcpf(Lacc[r]);
    const int sr = q0 + r + lhalf * 8;
    const size_t base = ((size_t)bI * Sc + sr) * Ec + h * HDc + nlo;
#pragma unroll
    for (int dt = 0; dt < 4; dt += 2) {
      const u32 pk = pack2(O[dt][r] * inv, O[dt + 1][r] * inv);
      *(u16*)&X[base + dt * 16]        = (u16)pk;
      *(u16*)&X[base + (dt + 1) * 16]  = (u16)(pk >> 16);
    }
  }
}

// ---------------------------------------------------------------------------
extern "C" void kernel_launch(void* const* d_in, const int* in_sizes, int n_in,
                              void* d_out, int out_size, void* d_ws, size_t ws_size,
                              hipStream_t stream) {
  const float* query = (const float*)d_in[0];
  const float* key_  = (const float*)d_in[1];
  const float* value = (const float*)d_in[2];
  const float* Wq = (const float*)d_in[3];
  const float* bq = (const float*)d_in[4];
  const float* Wk = (const float*)d_in[5];
  const float* bk = (const float*)d_in[6];
  const float* Wv = (const float*)d_in[7];
  const float* bv = (const float*)d_in[8];
  const float* Wo = (const float*)d_in[9];
  const float* bo = (const float*)d_in[10];

  char* ws = (char*)d_ws;
  const size_t WTB = (size_t)Ec * Ec * 2;     // 2 MiB per transposed weight
  const size_t HB  = (size_t)Mtot * Ec * 2;   // 16 MiB per bf16 activation
  bf16* wtq = (bf16*)(ws);
  bf16* wtk = (bf16*)(ws + WTB);
  bf16* wtv = (bf16*)(ws + 2 * WTB);
  bf16* wto = (bf16*)(ws + 3 * WTB);
  bf16* Qb  = (bf16*)(ws + 4 * WTB);
  bf16* Kb  = (bf16*)(ws + 4 * WTB + HB);
  bf16* Vtb = (bf16*)(ws + 4 * WTB + 2 * HB);
  bf16* Xb  = (bf16*)(ws + 4 * WTB + 3 * HB); // total ~75.5 MiB

  dim3 tb(32, 8, 1), tg(32, 32, 1);
  wconv_t<<<tg, tb, 0, stream>>>(Wq, wtq);
  wconv_t<<<tg, tb, 0, stream>>>(Wk, wtk);
  wconv_t<<<tg, tb, 0, stream>>>(Wv, wtv);
  wconv_t<<<tg, tb, 0, stream>>>(Wo, wto);

  dim3 gg(Ec / 128, Mtot / 128, 1);
  const float qscale = 0.125f * 1.44269504088896340736f; // 1/sqrt(HD) * log2(e)
  gemm_qkv<0, true><<<gg, 256, 0, stream>>>(query, wtq, bq, Qb, qscale);
  gemm_qkv<0, true><<<gg, 256, 0, stream>>>(key_,  wtk, bk, Kb, 1.0f);
  gemm_qkv<1, true><<<gg, 256, 0, stream>>>(value, wtv, bv, Vtb, 1.0f);

  attn_fwd<<<Bc * Hc * (Sc / 128), 256, 0, stream>>>(Qb, Kb, Vtb, Xb);

  gemm_qkv<2, false><<<gg, 256, 0, stream>>>(Xb, wto, bo, d_out, 1.0f);
}
